// UniformEntropicOTDistance_15934328668502
// MI455X (gfx1250) — matrix operand order repaired and checked by reading, not compiled, not run on hardware
//
#include <hip/hip_runtime.h>

// ---------------------------------------------------------------------------
// UniformEntropicOTDistance for MI455X (gfx1250, wave32, WMMA)
// B=64 batches, n=m=1024 points, d=64. One workgroup per batch; S/T tiles
// live in LDS (320KB/WGP) for all 50 Sinkhorn iterations; Gram matrix is
// recomputed each half-iteration with v_wmma_f32_16x16x32_f16.
// 512 threads = 16 waves; each wave owns 4 row tiles and shares each
// B-fragment load across them. The whole Sinkhorn runs in the log2 domain so
// the softmax uses raw v_exp_f32/v_log_f32 (no ln<->log2 multiplies).
// ---------------------------------------------------------------------------

typedef __attribute__((ext_vector_type(16))) _Float16 v16h;
typedef __attribute__((ext_vector_type(8)))  _Float16 v8h_t;
typedef __attribute__((ext_vector_type(8)))  float    v8f;

#define BATCH   64
#define N_PTS   1024
#define DIMS    64
#define ROW_PAD 72                       // halves per padded LDS row (144B, 16B aligned)
#define ITERS   50
#define NTHREADS 512

// log2-domain constants:  K2 = -(cost/eps)*log2(e) = min(K2COEF*dot - K2COEF, 0)
// with cost = max(2-2*dot,0), eps = 0.1.
#define K2COEF    28.853900817779268f    // 20 * log2(e)
#define N10LOG2E  (-14.426950408889634f) // -10 * log2(e)
#define LOG2_AB   (-10.0f)               // log2(1/1024)

// LDS layout (bytes)
#define SH_OFF   0
#define TH_OFF   (N_PTS * ROW_PAD * 2)               // 147456
#define LU_OFF   (2 * N_PTS * ROW_PAD * 2)           // 294912
#define LV_OFF   (LU_OFF + N_PTS * 4)                // 299008
#define RED_OFF  (LV_OFF + N_PTS * 4)                // 303104
#define SMEM_BYTES (RED_OFF + 64 * 4)                // 303360 <= 327680

// ---------------------------------------------------------------------------
__device__ __forceinline__ float fast_exp2(float x) {
  return __builtin_amdgcn_exp2f(x);      // v_exp_f32 (native base-2)
}
__device__ __forceinline__ float fast_log2(float x) {
  return __builtin_amdgcn_logf(x);       // v_log_f32 (native base-2)
}

__device__ __forceinline__ v8f wmma_f16(v16h a, v16h b, v8f c) {
  // (neg_a, A, neg_b, B, c_mod, C, reuse_a, reuse_b)
  return __builtin_amdgcn_wmma_f32_16x16x32_f16(false, a, false, b,
                                                (short)0, c, false, false);
}

// A-matrix 16x32 f16 fragment (ISA 7.12.2): lane l holds row M=l%16,
// K = {off..off+7, 16+off..16+off+7}, off = 8*(l/16).
__device__ __forceinline__ v16h load_fragA(const _Float16* m, int rowBase,
                                           int lane, int k0) {
  const _Float16* p = m + (size_t)(rowBase + (lane & 15)) * ROW_PAD
                        + k0 + ((lane >> 4) << 3);
  union { v16h v; v8h_t h[2]; } u;
  u.h[0] = *(const v8h_t*)(p);
  u.h[1] = *(const v8h_t*)(p + 16);
  return u.v;
}

// B-matrix 32x16 f16 fragment: lane l holds column N=l%16 with
// K = {off..off+7, 16+off..16+off+7}, off = 8*(l/16). B = T^T: column N of B is row N
// of T (contiguous in LDS).
__device__ __forceinline__ v16h load_fragB(const _Float16* m, int colBase,
                                           int lane, int k0) {
  const _Float16* p = m + (size_t)(colBase + (lane & 15)) * ROW_PAD
                        + k0 + ((lane >> 4) << 3);
  union { v16h v; v8h_t h[2]; } u;
  u.h[0] = *(const v8h_t*)(p);
  u.h[1] = *(const v8h_t*)(p + 16);
  return u.v;
}

// ---------------------------------------------------------------------------
// Block reductions for 512 threads (16 waves).
__device__ __forceinline__ float block_reduce_max(float v, float* red, int tid) {
  #pragma unroll
  for (int s = 16; s > 0; s >>= 1) v = fmaxf(v, __shfl_xor(v, s, 32));
  if ((tid & 31) == 0) red[tid >> 5] = v;
  __syncthreads();
  if (tid < 16) {
    float w = red[tid];
    #pragma unroll
    for (int s = 8; s > 0; s >>= 1) w = fmaxf(w, __shfl_xor(w, s, 32));
    if (tid == 0) red[16] = w;
  }
  __syncthreads();
  float r = red[16];
  __syncthreads();
  return r;
}

__device__ __forceinline__ float block_reduce_sum(float v, float* red, int tid) {
  #pragma unroll
  for (int s = 16; s > 0; s >>= 1) v += __shfl_xor(v, s, 32);
  if ((tid & 31) == 0) red[tid >> 5] = v;
  __syncthreads();
  float r = 0.0f;
  if (tid < 16) {
    float w = red[tid];
    #pragma unroll
    for (int s = 8; s > 0; s >>= 1) w += __shfl_xor(w, s, 32);
    if (tid == 0) red[16] = w;
  }
  __syncthreads();
  r = red[16];
  __syncthreads();
  return r;
}

// ---------------------------------------------------------------------------
// One Sinkhorn half-update in the log2 domain:
//   vout[r] = -10 - log2sumexp2_c( min(K2COEF*dot(A_r,B_c) - K2COEF, 0) + vin[c] )
// Stabilized with Mv = max(vin). Each wave owns 4 row tiles (16 waves * 4 =
// 64) and shares each B-fragment load across all 4 while sweeping the 64
// column tiles. All 4 WMMA chains complete before their exp2 epilogues so
// independent WMMAs can co-execute with the VALU epilogue.
__device__ __forceinline__ void sinkhorn_half(const _Float16* Am,
                                              const _Float16* Bm,
                                              const float* vin, float* vout,
                                              float* red, int tid) {
  float Mv = block_reduce_max(fmaxf(vin[tid], vin[tid + NTHREADS]), red, tid);
  int wave = tid >> 5, lane = tid & 31;
  int rowBase = wave * 64;

  v16h a[4][2];
  #pragma unroll
  for (int rt = 0; rt < 4; ++rt) {
    a[rt][0] = load_fragA(Am, rowBase + rt * 16, lane, 0);
    a[rt][1] = load_fragA(Am, rowBase + rt * 16, lane, 32);
  }
  float acc[4][8];
  #pragma unroll
  for (int rt = 0; rt < 4; ++rt)
    #pragma unroll
    for (int i = 0; i < 8; ++i) acc[rt][i] = 0.0f;

  #pragma unroll 1
  for (int ct = 0; ct < 64; ++ct) {
    v16h b0 = load_fragB(Bm, ct * 16, lane, 0);
    v16h b1 = load_fragB(Bm, ct * 16, lane, 32);
    float lvv = vin[ct * 16 + (lane & 15)] - Mv;

    v8f cM[4];
    #pragma unroll
    for (int rt = 0; rt < 4; ++rt) {
      v8f c = {};
      c = wmma_f16(a[rt][0], b0, c);
      cM[rt] = wmma_f16(a[rt][1], b1, c);
    }
    #pragma unroll
    for (int rt = 0; rt < 4; ++rt) {
      #pragma unroll
      for (int i = 0; i < 8; ++i) {
        float arg = fminf(fmaf(K2COEF, cM[rt][i], -K2COEF), 0.0f) + lvv;
        acc[rt][i] += fast_exp2(arg);
      }
    }
  }
  // sum over the 16 lanes sharing each row (C layout: col = lane%16)
  #pragma unroll
  for (int rt = 0; rt < 4; ++rt) {
    #pragma unroll
    for (int i = 0; i < 8; ++i) {
      #pragma unroll
      for (int s = 1; s < 16; s <<= 1) acc[rt][i] += __shfl_xor(acc[rt][i], s, 32);
    }
    if ((lane & 15) == 0) {
      #pragma unroll
      for (int i = 0; i < 8; ++i) {
        int row = rowBase + rt * 16 + i + ((lane >> 4) << 3);
        vout[row] = LOG2_AB - Mv - fast_log2(acc[rt][i]);
      }
    }
  }
}

// ---------------------------------------------------------------------------
// Kernel 1: L2-normalize rows, convert to f16. One wave per 64-dim row.
__global__ __launch_bounds__(256)
void prep_kernel(const float* __restrict__ src, const float* __restrict__ tgt,
                 _Float16* __restrict__ Sg, _Float16* __restrict__ Tg) {
  int gid  = blockIdx.x * blockDim.x + threadIdx.x;
  int w    = gid >> 5;
  int lane = gid & 31;
  const float* rp;
  _Float16* wp;
  if (w < BATCH * N_PTS) {
    rp = src + (size_t)w * DIMS;
    wp = Sg + (size_t)w * DIMS;
  } else {
    int r = w - BATCH * N_PTS;
    rp = tgt + (size_t)r * DIMS;
    wp = Tg + (size_t)r * DIMS;
  }
  float2 x = ((const float2*)rp)[lane];
  float ss = x.x * x.x + x.y * x.y;
  #pragma unroll
  for (int s = 16; s > 0; s >>= 1) ss += __shfl_xor(ss, s, 32);
  float inv = 1.0f / fmaxf(sqrtf(ss), 1e-12f);
  union { _Float16 h[2]; unsigned u; } pk;
  pk.h[0] = (_Float16)(x.x * inv);
  pk.h[1] = (_Float16)(x.y * inv);
  ((unsigned*)wp)[lane] = pk.u;
}

// ---------------------------------------------------------------------------
// Kernel 2: one workgroup per batch; S/T resident in LDS; 50 Sinkhorn
// iterations + transport-cost epilogue.
__global__ __launch_bounds__(NTHREADS)
void sinkhorn_kernel(const _Float16* __restrict__ Sg,
                     const _Float16* __restrict__ Tg,
                     float* __restrict__ dist) {
  extern __shared__ char smem[];
  _Float16* Sh = (_Float16*)(smem + SH_OFF);
  _Float16* Th = (_Float16*)(smem + TH_OFF);
  float* lu  = (float*)(smem + LU_OFF);
  float* lv  = (float*)(smem + LV_OFF);
  float* red = (float*)(smem + RED_OFF);

  int tid = threadIdx.x;
  int b   = blockIdx.x;

  // Stage both point sets into padded LDS (2 rows per thread, 8x16B each).
  #pragma unroll
  for (int rr = 0; rr < 2; ++rr) {
    int r = tid + rr * NTHREADS;
    const uint4* sg = (const uint4*)(Sg + (size_t)b * N_PTS * DIMS + (size_t)r * DIMS);
    const uint4* tg = (const uint4*)(Tg + (size_t)b * N_PTS * DIMS + (size_t)r * DIMS);
    uint4* sd = (uint4*)(Sh + (size_t)r * ROW_PAD);
    uint4* td = (uint4*)(Th + (size_t)r * ROW_PAD);
    #pragma unroll
    for (int i = 0; i < 8; ++i) { sd[i] = sg[i]; td[i] = tg[i]; }
    lu[r] = 0.0f;
    lv[r] = 0.0f;
  }
  __syncthreads();

  for (int it = 0; it < ITERS; ++it) {
    sinkhorn_half(Sh, Th, lv, lu, red, tid);   // u-update: rows of S vs cols of T
    __syncthreads();
    sinkhorn_half(Th, Sh, lu, lv, red, tid);   // v-update: rows of T vs cols of S
    __syncthreads();
  }

  // Epilogue: sum of plan*cost = exp2(-10*log2e*cost + lu2[n] + lv2[m]) * cost
  int wave = tid >> 5, lane = tid & 31;
  int rowBase = wave * 64;
  float acc = 0.0f;

  v16h a[4][2];
  float lur[4][8];
  #pragma unroll
  for (int rt = 0; rt < 4; ++rt) {
    a[rt][0] = load_fragA(Sh, rowBase + rt * 16, lane, 0);
    a[rt][1] = load_fragA(Sh, rowBase + rt * 16, lane, 32);
    #pragma unroll
    for (int i = 0; i < 8; ++i)
      lur[rt][i] = lu[rowBase + rt * 16 + i + ((lane >> 4) << 3)];
  }
  #pragma unroll 1
  for (int ct = 0; ct < 64; ++ct) {
    v16h b0 = load_fragB(Th, ct * 16, lane, 0);
    v16h b1 = load_fragB(Th, ct * 16, lane, 32);
    float lvv = lv[ct * 16 + (lane & 15)];

    v8f cM[4];
    #pragma unroll
    for (int rt = 0; rt < 4; ++rt) {
      v8f c = {};
      c = wmma_f16(a[rt][0], b0, c);
      cM[rt] = wmma_f16(a[rt][1], b1, c);
    }
    #pragma unroll
    for (int rt = 0; rt < 4; ++rt) {
      #pragma unroll
      for (int i = 0; i < 8; ++i) {
        float cost = fmaxf(fmaf(-2.0f, cM[rt][i], 2.0f), 0.0f);
        float arg  = fmaf(N10LOG2E, cost, lur[rt][i] + lvv);
        acc = fmaf(fast_exp2(arg), cost, acc);
      }
    }
  }
  float tot = block_reduce_sum(acc, red, tid);
  if (tid == 0) dist[b] = tot;
}

// ---------------------------------------------------------------------------
// Kernel 3: mean over batch -> scalar output.
__global__ __launch_bounds__(32)
void mean_kernel(const float* __restrict__ dist, float* __restrict__ out) {
  int lane = threadIdx.x;
  float v = dist[lane] + dist[lane + 32];
  #pragma unroll
  for (int s = 16; s > 0; s >>= 1) v += __shfl_xor(v, s, 32);
  if (lane == 0) out[0] = v * (1.0f / (float)BATCH);
}

// ---------------------------------------------------------------------------
extern "C" void kernel_launch(void* const* d_in, const int* in_sizes, int n_in,
                              void* d_out, int out_size, void* d_ws, size_t ws_size,
                              hipStream_t stream) {
  (void)in_sizes; (void)n_in; (void)out_size; (void)ws_size;
  const float* src = (const float*)d_in[0];
  const float* tgt = (const float*)d_in[1];

  // Workspace: S_h (8MB f16) | T_h (8MB f16) | dist (64 f32)  ~= 16 MB
  char* ws = (char*)d_ws;
  const size_t tile_bytes = (size_t)BATCH * N_PTS * DIMS * sizeof(_Float16);
  _Float16* Sg = (_Float16*)ws;
  _Float16* Tg = (_Float16*)(ws + tile_bytes);
  float* dist  = (float*)(ws + 2 * tile_bytes);

  (void)hipFuncSetAttribute((const void*)sinkhorn_kernel,
                            hipFuncAttributeMaxDynamicSharedMemorySize,
                            SMEM_BYTES);

  // 2*64*1024 rows, one wave32 per row, 8 waves per block.
  prep_kernel<<<(2 * BATCH * N_PTS) / 8, 256, 0, stream>>>(src, tgt, Sg, Tg);
  sinkhorn_kernel<<<BATCH, NTHREADS, SMEM_BYTES, stream>>>(Sg, Tg, dist);
  mean_kernel<<<1, 32, 0, stream>>>(dist, (float*)d_out);
}
